// ouroboros_35562329211547
// MI455X (gfx1250) — matrix operand order repaired and checked by reading, not compiled, not run on hardware
//
#include <hip/hip_runtime.h>
#include <stdint.h>

#define BB 4
#define LL 2048
#define DD 1024
#define MTOK (BB*LL)   // 8192 tokens

// GEMM tiling
#define BM 128
#define BN 128
#define BK 32
#define LDSS 40        // LDS row stride in halves (32 + 8 pad -> spreads banks)

typedef float v8f  __attribute__((ext_vector_type(8)));
typedef __bf16 v16bf __attribute__((ext_vector_type(16)));

union FragU { v16bf v; uint4 q[2]; };

__device__ __forceinline__ unsigned short f2bf(float f) {
  unsigned int u = __float_as_uint(f);
  unsigned int r = u + 0x7FFFu + ((u >> 16) & 1u);   // round-to-nearest-even
  return (unsigned short)(r >> 16);
}
__device__ __forceinline__ float sigm(float x) { return 1.f / (1.f + __expf(-x)); }

// ---------------------------------------------------------------------------
// WMMA GEMM:  C[M,N] = act( A[M,K](bf16) * W[N,K]^T(bf16) + bias ) + residual
// Block = 256 thr (8 waves) -> 128x128 C tile. Wave = 32x64 via 2x4
// v_wmma_f32_16x16x32_bf16 per K-step. A/B K-chunks double-buffered in LDS.
// act: 0=none 1=silu 2=softplus
// ---------------------------------------------------------------------------
__global__ __launch_bounds__(256) void wmma_gemm_bf16(
    const unsigned short* __restrict__ A,
    const unsigned short* __restrict__ W,
    const float* __restrict__ bias,
    const float* __restrict__ residual,
    float* __restrict__ C,
    int M, int N, int K, int act)
{
  __shared__ unsigned short lA[2][BM * LDSS];
  __shared__ unsigned short lB[2][BN * LDSS];

  const int tid  = threadIdx.x;
  const int lane = tid & 31;
  const int wid  = tid >> 5;
  const int wm   = wid & 3;          // 4 waves along M  (32 rows each)
  const int wn   = wid >> 2;         // 2 waves along N  (64 cols each)
  const int hlf  = lane >> 4;
  const int lm   = lane & 15;

  const int blockM = blockIdx.y * BM;
  const int blockN = blockIdx.x * BN;

  // cooperative fill: 256 threads x 2 chunks x 8 halves cover 128x32 tile
  const int q0 = tid * 2, q1 = tid * 2 + 1;
  const int ar0 = q0 >> 2, ac0 = (q0 & 3) * 8;
  const int ar1 = q1 >> 2, ac1 = (q1 & 3) * 8;

  const unsigned short* gA0 = A + (size_t)(blockM + ar0) * K + ac0;
  const unsigned short* gA1 = A + (size_t)(blockM + ar1) * K + ac1;
  int wr0 = blockN + ar0; if (wr0 >= N) wr0 = N - 1;
  int wr1 = blockN + ar1; if (wr1 >= N) wr1 = N - 1;
  const unsigned short* gW0 = W + (size_t)wr0 * K + ac0;
  const unsigned short* gW1 = W + (size_t)wr1 * K + ac1;

  v8f acc[2][4] = {};

  // prologue: fetch K-chunk 0 and park it in LDS buffer 0
  uint4 ra0 = *(const uint4*)(gA0);
  uint4 ra1 = *(const uint4*)(gA1);
  uint4 rb0 = *(const uint4*)(gW0);
  uint4 rb1 = *(const uint4*)(gW1);
  *(uint4*)&lA[0][ar0 * LDSS + ac0] = ra0;
  *(uint4*)&lA[0][ar1 * LDSS + ac1] = ra1;
  *(uint4*)&lB[0][ar0 * LDSS + ac0] = rb0;
  *(uint4*)&lB[0][ar1 * LDSS + ac1] = rb1;

  int buf = 0;
  for (int k0 = 0; k0 < K; k0 += BK) {
    __syncthreads();
    const bool more = (k0 + BK) < K;
    if (more) {
      ra0 = *(const uint4*)(gA0 + k0 + BK);
      ra1 = *(const uint4*)(gA1 + k0 + BK);
      rb0 = *(const uint4*)(gW0 + k0 + BK);
      rb1 = *(const uint4*)(gW1 + k0 + BK);
      if (k0 + 2 * BK < K) {
        __builtin_prefetch((const void*)(gA0 + k0 + 2 * BK), 0, 3);
        __builtin_prefetch((const void*)(gW0 + k0 + 2 * BK), 0, 3);
      }
    }

    // A fragments (16x32 layout): elems 0..7 -> K=8*hlf.. ; 8..15 -> K=16+8*hlf..
    FragU a[2], b[4];
#pragma unroll
    for (int i = 0; i < 2; i++) {
      const unsigned short* p = &lA[buf][(wm * 32 + i * 16 + lm) * LDSS];
      a[i].q[0] = *(const uint4*)(p + 8 * hlf);
      a[i].q[1] = *(const uint4*)(p + 16 + 8 * hlf);
    }
    // B fragments (32x16 layout): lane=col, elems 0..7 -> K=8*hlf.. ; 8..15 -> K=16+8*hlf..
#pragma unroll
    for (int j = 0; j < 4; j++) {
      const unsigned short* p = &lB[buf][(wn * 64 + j * 16 + lm) * LDSS];
      b[j].q[0] = *(const uint4*)(p + 8 * hlf);
      b[j].q[1] = *(const uint4*)(p + 16 + 8 * hlf);
    }

#pragma unroll
    for (int i = 0; i < 2; i++)
#pragma unroll
      for (int j = 0; j < 4; j++)
        acc[i][j] = __builtin_amdgcn_wmma_f32_16x16x32_bf16(
            false, a[i].v, false, b[j].v, (short)0, acc[i][j], false, false);

    if (more) {
      const int nb = buf ^ 1;
      *(uint4*)&lA[nb][ar0 * LDSS + ac0] = ra0;
      *(uint4*)&lA[nb][ar1 * LDSS + ac1] = ra1;
      *(uint4*)&lB[nb][ar0 * LDSS + ac0] = rb0;
      *(uint4*)&lB[nb][ar1 * LDSS + ac1] = rb1;
    }
    buf ^= 1;
  }

  // Epilogue: D elem r -> row blockM + wm*32 + i*16 + r + 8*hlf,
  //           col = blockN + wn*64 + j*16 + lm
#pragma unroll
  for (int j = 0; j < 4; j++) {
    int col = blockN + wn * 64 + j * 16 + lm;
    if (col >= N) continue;
    float bv = bias ? bias[col] : 0.f;
#pragma unroll
    for (int i = 0; i < 2; i++) {
#pragma unroll
      for (int r = 0; r < 8; r++) {
        int row = blockM + wm * 32 + i * 16 + r + 8 * hlf;
        if (row >= M) continue;
        float v = acc[i][j][r] + bv;
        if (act == 1)      v = v * sigm(v);
        else if (act == 2) v = (v > 20.f) ? v : log1pf(__expf(v));
        size_t o = (size_t)row * N + col;
        if (residual) v += residual[o];
        C[o] = v;
      }
    }
  }
}

// ---------------------------------------------------------------------------
// u = flip(concat([y-x, y], -1), axis=L)   -> (B,L,1024) f32
// ---------------------------------------------------------------------------
__global__ void prep_u(const float* __restrict__ x, const float* __restrict__ y,
                       float* __restrict__ u)
{
  int idx = blockIdx.x * blockDim.x + threadIdx.x;       // m*1024 + c
  int c = idx & (DD - 1);
  int m = idx >> 10;
  int t = m & (LL - 1);
  int b = m >> 11;
  int ts = LL - 1 - t;
  size_t so = ((size_t)(b * LL + ts)) * 512;
  u[idx] = (c < 512) ? (y[so + c] - x[so + c]) : y[so + (c - 512)];
}

// rmsnorm over 1024 -> bf16 out. one block (256 thr) per token.
__global__ __launch_bounds__(256) void rmsnorm_to_bf16(
    const float* __restrict__ x, const float* __restrict__ w,
    unsigned short* __restrict__ out)
{
  __shared__ float red[8];
  __shared__ float scale;
  int tok = blockIdx.x;
  const float* xr = x + (size_t)tok * DD;
  float v[4]; float s = 0.f;
#pragma unroll
  for (int i = 0; i < 4; i++) { v[i] = xr[threadIdx.x + 256 * i]; s += v[i] * v[i]; }
  for (int off = 16; off; off >>= 1) s += __shfl_xor(s, off, 32);
  if ((threadIdx.x & 31) == 0) red[threadIdx.x >> 5] = s;
  __syncthreads();
  if (threadIdx.x == 0) {
    float t = 0.f;
    for (int i = 0; i < 8; i++) t += red[i];
    scale = rsqrtf(t / (float)DD + 1e-5f);
  }
  __syncthreads();
  unsigned short* o = out + (size_t)tok * DD;
  float sc = scale;
#pragma unroll
  for (int i = 0; i < 4; i++) {
    int c = threadIdx.x + 256 * i;
    o[c] = f2bf(v[i] * sc * w[c]);
  }
}

// causal depthwise conv (d_conv=4) on xz[:, :1024] + SiLU -> f32 and bf16
__global__ void conv_silu(const float* __restrict__ xz, const float* __restrict__ cw,
                          const float* __restrict__ cb,
                          float* __restrict__ xf, unsigned short* __restrict__ xb)
{
  int idx = blockIdx.x * blockDim.x + threadIdx.x;       // m*1024 + e
  int e = idx & (DD - 1);
  int m = idx >> 10;
  int t = m & (LL - 1);
  int b = m >> 11;
  float acc = cb[e];
#pragma unroll
  for (int j = 0; j < 4; j++) {
    int tt = t - 3 + j;
    if (tt >= 0) acc += cw[e * 4 + j] * xz[((size_t)(b * LL + tt)) * 2048 + e];
  }
  acc = acc * sigm(acc);
  xf[idx] = acc;
  xb[idx] = f2bf(acc);
}

__global__ void to_bf16(const float* __restrict__ s, unsigned short* __restrict__ d, int n)
{
  int i = blockIdx.x * blockDim.x + threadIdx.x;
  if (i < n) d[i] = f2bf(s[i]);
}

// strided f32 -> compact bf16 (for dbc[:, :64])
__global__ void strided_to_bf16(const float* __restrict__ s, unsigned short* __restrict__ d,
                                int cols, int ld)
{
  int i = blockIdx.x * blockDim.x + threadIdx.x;
  int m = i / cols, c = i - m * cols;
  d[i] = f2bf(s[(size_t)m * ld + c]);
}

// selective scan: thread = (batch, channel); 16 states in regs; sequential in L
__global__ __launch_bounds__(256) void selective_scan(
    const float* __restrict__ u, const float* __restrict__ delta,
    const float* __restrict__ dbc, const float* __restrict__ Alog,
    const float* __restrict__ Dv, float* __restrict__ ys)
{
  int idx = blockIdx.x * 256 + threadIdx.x;              // b*1024 + e
  int e = idx & (DD - 1);
  int b = idx >> 10;
  float Ae[16], h[16];
#pragma unroll
  for (int n = 0; n < 16; n++) { Ae[n] = -__expf(Alog[e * 16 + n]); h[n] = 0.f; }
  float Dval = Dv[e];
  const float* ub = u     + (size_t)b * LL * DD;
  const float* db = delta + (size_t)b * LL * DD;
  const float* bc = dbc   + (size_t)b * LL * 96;
  float* yb = ys + (size_t)b * LL * DD;
  for (int t = 0; t < LL; t++) {
    float dt = db[(size_t)t * DD + e];
    float ut = ub[(size_t)t * DD + e];
    const float* p = bc + (size_t)t * 96;                // wave-uniform base
    float du = dt * ut, y = 0.f;
#pragma unroll
    for (int n = 0; n < 16; n++) {
      float dA = __expf(dt * Ae[n]);
      h[n] = dA * h[n] + du * p[64 + n];
      y += h[n] * p[80 + n];
    }
    yb[(size_t)t * DD + e] = y + ut * Dval;
  }
}

// g = ys * silu(z)  (z = xz[:, 1024:2048]) -> bf16
__global__ void gate_bf16(const float* __restrict__ ys, const float* __restrict__ xz,
                          unsigned short* __restrict__ g)
{
  int idx = blockIdx.x * blockDim.x + threadIdx.x;
  int e = idx & (DD - 1);
  int m = idx >> 10;
  float z = xz[(size_t)m * 2048 + 1024 + e];
  g[idx] = f2bf(ys[idx] * z * sigm(z));
}

// state_pred = flip(splin, axis=L), stored to output AND to data-mamba input
__global__ void flip_store(const float* __restrict__ sp, float* __restrict__ osp,
                           float* __restrict__ cur)
{
  int idx = blockIdx.x * blockDim.x + threadIdx.x;
  int c = idx & (DD - 1);
  int m = idx >> 10;
  int t = m & (LL - 1);
  int b = m >> 11;
  float v = sp[((size_t)(b * LL + (LL - 1 - t))) * DD + c];
  osp[idx] = v;
  cur[idx] = v;
}

// yhat = cur @ out_W.T + out_b   (N=1). one wave per token.
__global__ __launch_bounds__(256) void out_dot(const float* __restrict__ cur,
                                               const float* __restrict__ w,
                                               const float* __restrict__ bias,
                                               float* __restrict__ yhat)
{
  int wid = threadIdx.x >> 5, lane = threadIdx.x & 31;
  int tok = blockIdx.x * 8 + wid;
  const float* r = cur + (size_t)tok * DD;
  float s = 0.f;
  for (int c = lane; c < DD; c += 32) s += r[c] * w[c];
  for (int off = 16; off; off >>= 1) s += __shfl_xor(s, off, 32);
  if (lane == 0) yhat[tok] = s + bias[0];
}

// ---------------------------------------------------------------------------
extern "C" void kernel_launch(void* const* d_in, const int* in_sizes, int n_in,
                              void* d_out, int out_size, void* d_ws, size_t ws_size,
                              hipStream_t stream)
{
  (void)in_sizes; (void)n_in; (void)out_size; (void)ws_size;
  // input order: x, y, control_params(2 x 10), data_params(2 x 10),
  //              cp_W, cp_b, out_W, out_b, mask_ind (= -1, unused)
  const float* x    = (const float*)d_in[0];
  const float* y    = (const float*)d_in[1];
  const float* cp_W = (const float*)d_in[42];
  const float* cp_b = (const float*)d_in[43];
  const float* oW   = (const float*)d_in[44];
  const float* ob   = (const float*)d_in[45];

  // workspace bump allocator
  char* base = (char*)d_ws; size_t off = 0;
  auto alloc = [&](size_t bytes) -> void* {
    void* p = base + off; off += (bytes + 255) & ~(size_t)255; return p;
  };
  float*          cur   = (float*)alloc((size_t)MTOK * DD * 4);
  float*          nxt   = (float*)alloc((size_t)MTOK * DD * 4);
  unsigned short* abf   = (unsigned short*)alloc((size_t)MTOK * DD * 2);
  float*          xz    = (float*)alloc((size_t)MTOK * 2048 * 4);
  float*          xiA   = (float*)alloc((size_t)MTOK * DD * 4);
  unsigned short* xibf  = (unsigned short*)alloc((size_t)MTOK * DD * 2);
  float*          dbc   = (float*)alloc((size_t)MTOK * 96 * 4);
  unsigned short* dtbf  = (unsigned short*)alloc((size_t)MTOK * 64 * 2);
  float*          delta = (float*)alloc((size_t)MTOK * DD * 4);
  float*          ysb   = (float*)alloc((size_t)MTOK * DD * 4);
  unsigned short* gbf   = (unsigned short*)alloc((size_t)MTOK * DD * 2);
  unsigned short* wbf   = (unsigned short*)alloc((size_t)2048 * 1024 * 2);
  float*          splin = (float*)alloc((size_t)MTOK * DD * 4);

  const int EW = (MTOK * DD) / 256;                 // elementwise grid (32768)
  dim3 gN2048(2048 / BN, MTOK / BM);
  dim3 gN1024(1024 / BN, MTOK / BM);
  dim3 gN96((96 + BN - 1) / BN, MTOK / BM);

  auto mamba_layer = [&](int pbase) {
    const float* norm_w  = (const float*)d_in[pbase + 0];
    const float* in_proj = (const float*)d_in[pbase + 1];
    const float* conv_w  = (const float*)d_in[pbase + 2];
    const float* conv_b  = (const float*)d_in[pbase + 3];
    const float* x_proj  = (const float*)d_in[pbase + 4];
    const float* dt_proj = (const float*)d_in[pbase + 5];
    const float* dt_b    = (const float*)d_in[pbase + 6];
    const float* A_log   = (const float*)d_in[pbase + 7];
    const float* Dv      = (const float*)d_in[pbase + 8];
    const float* o_proj  = (const float*)d_in[pbase + 9];

    rmsnorm_to_bf16<<<MTOK, 256, 0, stream>>>(cur, norm_w, abf);
    to_bf16<<<(2048 * 1024) / 256, 256, 0, stream>>>(in_proj, wbf, 2048 * 1024);
    wmma_gemm_bf16<<<gN2048, 256, 0, stream>>>(abf, wbf, nullptr, nullptr, xz,
                                               MTOK, 2048, 1024, 0);
    conv_silu<<<EW, 256, 0, stream>>>(xz, conv_w, conv_b, xiA, xibf);
    to_bf16<<<(96 * 1024) / 256, 256, 0, stream>>>(x_proj, wbf, 96 * 1024);
    wmma_gemm_bf16<<<gN96, 256, 0, stream>>>(xibf, wbf, nullptr, nullptr, dbc,
                                             MTOK, 96, 1024, 0);
    strided_to_bf16<<<(MTOK * 64) / 256, 256, 0, stream>>>(dbc, dtbf, 64, 96);
    to_bf16<<<(1024 * 64) / 256, 256, 0, stream>>>(dt_proj, wbf, 1024 * 64);
    wmma_gemm_bf16<<<gN1024, 256, 0, stream>>>(dtbf, wbf, dt_b, nullptr, delta,
                                               MTOK, 1024, 64, 2 /*softplus*/);
    selective_scan<<<(BB * DD) / 256, 256, 0, stream>>>(xiA, delta, dbc, A_log, Dv, ysb);
    gate_bf16<<<EW, 256, 0, stream>>>(ysb, xz, gbf);
    to_bf16<<<(1024 * 1024) / 256, 256, 0, stream>>>(o_proj, wbf, 1024 * 1024);
    wmma_gemm_bf16<<<gN1024, 256, 0, stream>>>(gbf, wbf, nullptr, cur, nxt,
                                               MTOK, 1024, 1024, 0);
    float* t = cur; cur = nxt; nxt = t;
  };

  // control stack on u = flip(concat([y-x, y]))
  prep_u<<<EW, 256, 0, stream>>>(x, y, cur);
  mamba_layer(2);
  mamba_layer(12);

  // state_pred = flip(silu(sp @ cp_W.T + cp_b))
  to_bf16<<<EW, 256, 0, stream>>>(cur, abf, MTOK * DD);
  to_bf16<<<(1024 * 1024) / 256, 256, 0, stream>>>(cp_W, wbf, 1024 * 1024);
  wmma_gemm_bf16<<<gN1024, 256, 0, stream>>>(abf, wbf, cp_b, nullptr, splin,
                                             MTOK, 1024, 1024, 1 /*silu*/);
  float* out_sp = (float*)d_out + MTOK;   // yhat first (8192 f32), then state_pred
  flip_store<<<EW, 256, 0, stream>>>(splin, out_sp, cur);

  // data stack on state_pred
  mamba_layer(22);
  mamba_layer(32);

  // yhat = cur @ out_W.T + out_b
  out_dot<<<MTOK / 8, 256, 0, stream>>>(cur, oW, ob, (float*)d_out);
}
